// Attention_41085657153633
// MI455X (gfx1250) — compile-verified, matrix operand order checked by reading, not run on hardware
//
#include <hip/hip_runtime.h>

typedef __attribute__((ext_vector_type(16))) _Float16 v16h;
typedef __attribute__((ext_vector_type(8)))  _Float16 v8h;
typedef __attribute__((ext_vector_type(8)))  float    v8f;

// Gather a 16-bit A/B WMMA fragment: per CDNA5 ISA 7.12.2, lane L (half=L/16)
// holds K = half*8 + 0..7 (VGPR0..3) and 16 + half*8 + 0..7 (VGPR4..7).
// Those are two contiguous 16-byte chunks -> two ds_load_b128.
__device__ inline v16h ld_frag(const _Float16* p0, const _Float16* p1) {
  v8h a = *(const v8h*)p0;
  v8h b = *(const v8h*)p1;
  v16h r;
#pragma unroll
  for (int i = 0; i < 8; i++) { r[i] = a[i]; r[i + 8] = b[i]; }
  return r;
}

__device__ inline v8f wmma_f16(v16h a, v16h b, v8f c) {
  return __builtin_amdgcn_wmma_f32_16x16x32_f16(false, a, false, b, (short)0, c,
                                                false, false);
}

// CDNA5 async copy: GLOBAL_LOAD_ASYNC_TO_LDS_B128, GVS mode
// (addr = SGPR64 base + 32-bit per-lane byte offset), tracked by ASYNCcnt.
// LDS destination = low 32 bits of the generic shared-pointer (= LDS offset).
__device__ inline void async_b128(const _Float16* lds, const void* gbase,
                                  unsigned goff_bytes) {
  unsigned l = (unsigned)(unsigned long long)lds;
  asm volatile("global_load_async_to_lds_b128 %0, %1, %2 offset:0"
               :
               : "v"(l), "v"(goff_bytes), "s"((unsigned long long)gbase)
               : "memory");
}

__device__ inline void wait_async0() {
#if __has_builtin(__builtin_amdgcn_s_wait_asynccnt)
  __builtin_amdgcn_s_wait_asynccnt(0);
#else
  asm volatile("s_wait_asynccnt 0" ::: "memory");
#endif
}

// ---------------------------------------------------------------- convert
__global__ __launch_bounds__(256) void f32_to_f16(const float* __restrict__ in,
                                                  _Float16* __restrict__ out,
                                                  int n) {
  int i = blockIdx.x * 256 + threadIdx.x;
  if (i < n) out[i] = (_Float16)in[i];
}

// ------------------------------------------------------- QKV GEMM (WMMA)
// QKV[t, c] = sum_k X[t,k] * W[c,k];  X: 8192x384 f16, W: 1152x384 f16.
// Block tile 128(M) x 64(N), 8 waves in a 4x2 grid, each wave owns 2x2
// 16x16 tiles (4 WMMAs / 8 b128 LDS loads per K-step). K-step 32.
// A/B tiles double-buffered, staged with async global->LDS copies.
// Epilogue scatters to head-major q/k/v: col c -> (c/384, (c%384)/64, c%64).
__global__ __launch_bounds__(256) void gemm_qkv(const _Float16* __restrict__ X,
                                                const _Float16* __restrict__ W,
                                                _Float16* __restrict__ q,
                                                _Float16* __restrict__ k,
                                                _Float16* __restrict__ v) {
  const int Kd = 384;
  __shared__ __align__(16) _Float16 As[2][128 * 40];  // stride 40 f16 = 80 B
  __shared__ __align__(16) _Float16 Bs[2][64 * 40];
  const int tid = threadIdx.x, wave = tid >> 5, lane = tid & 31;
  const int half = lane >> 4, ml = lane & 15;
  const int mbase = blockIdx.y * 128, nbase = blockIdx.x * 64;
  const int mg = wave >> 1, ng = wave & 1;

  auto stage = [&](int buf, int kb) {
#pragma unroll
    for (int c = 0; c < 2; c++) {  // A: 128 rows x 4 segs = 512 b128 chunks
      const int idx = tid + c * 256;
      const int row = idx >> 2, seg = idx & 3;
      async_b128(&As[buf][row * 40 + seg * 8], X,
                 (unsigned)((((mbase + row) * Kd) + kb + seg * 8) * 2));
    }
    {  // B: 64 rows x 4 segs = 256 chunks
      const int row = tid >> 2, seg = tid & 3;
      async_b128(&Bs[buf][row * 40 + seg * 8], W,
                 (unsigned)((((nbase + row) * Kd) + kb + seg * 8) * 2));
    }
  };

  v8f acc00 = {}, acc01 = {}, acc10 = {}, acc11 = {};
  stage(0, 0);
  wait_async0();
  __syncthreads();
  int cur = 0;
  for (int kb = 0; kb < Kd; kb += 32) {
    if (kb + 32 < Kd) stage(cur ^ 1, kb + 32);  // prefetch next tile
    const _Float16* A0 = &As[cur][(mg * 32 + ml) * 40];
    const _Float16* A1 = &As[cur][(mg * 32 + 16 + ml) * 40];
    const _Float16* B0 = &Bs[cur][(ng * 32 + ml) * 40];
    const _Float16* B1 = &Bs[cur][(ng * 32 + 16 + ml) * 40];
    v16h a0 = ld_frag(A0 + half * 8, A0 + 16 + half * 8);
    v16h a1 = ld_frag(A1 + half * 8, A1 + 16 + half * 8);
    v16h b0 = ld_frag(B0 + half * 8, B0 + 16 + half * 8);
    v16h b1 = ld_frag(B1 + half * 8, B1 + 16 + half * 8);
    acc00 = wmma_f16(a0, b0, acc00);
    acc01 = wmma_f16(a0, b1, acc01);
    acc10 = wmma_f16(a1, b0, acc10);
    acc11 = wmma_f16(a1, b1, acc11);
    wait_async0();
    __syncthreads();
    cur ^= 1;
  }
#pragma unroll
  for (int mi = 0; mi < 2; mi++) {
#pragma unroll
    for (int ni = 0; ni < 2; ni++) {
      v8f acc = (mi == 0) ? (ni == 0 ? acc00 : acc01)
                          : (ni == 0 ? acc10 : acc11);
      const int gcol = nbase + (ng * 2 + ni) * 16 + ml;
      const int which = gcol / 384, rem = gcol % 384;
      const int h = rem / 64, d = rem % 64;
      _Float16* dst = (which == 0) ? q : ((which == 1) ? k : v);
#pragma unroll
      for (int r = 0; r < 8; r++) {
        const int tok = mbase + (mg * 2 + mi) * 16 + r + 8 * half;
        const int bb = tok >> 10, nn = tok & 1023;
        dst[(((size_t)bb * 6 + h) * 1024 + nn) * 64 + d] = (_Float16)acc[r];
      }
    }
  }
}

// ------------------------------------------- per-(token,head) normalization
// rows of 64; torch std ddof=1; optional extra scale folded in (q gets 1/8).
__global__ __launch_bounds__(256) void norm64(_Float16* __restrict__ buf,
                                              float extra_scale) {
  const int wave = threadIdx.x >> 5, lane = threadIdx.x & 31;
  const size_t row = (size_t)blockIdx.x * 8 + wave;
  _Float16* p = buf + row * 64 + lane * 2;
  float v0 = (float)p[0], v1 = (float)p[1];
  float s = v0 + v1, sq = v0 * v0 + v1 * v1;
#pragma unroll
  for (int off = 1; off < 32; off <<= 1) {
    s += __shfl_xor(s, off);
    sq += __shfl_xor(sq, off);
  }
  const float mean = s * (1.0f / 64.0f);
  const float var = (sq - 64.0f * mean * mean) * (1.0f / 63.0f);
  const float inv = rsqrtf(var) * extra_scale;
  p[0] = (_Float16)((v0 - mean) * inv);
  p[1] = (_Float16)((v1 - mean) * inv);
}

// ------------------------------------------------ flash attention (WMMA)
// grid: (8 query-blocks of 128, 48 bh). Block = 8 waves; wave owns a 16-row
// query tile. Key blocks of 32 -> P(16x32) is exactly one WMMA A operand.
// Q and K tiles staged with async global->LDS; V staged manually (transposed).
__global__ __launch_bounds__(256) void attn_fwd(const _Float16* __restrict__ Q,
                                                const _Float16* __restrict__ K,
                                                const _Float16* __restrict__ V,
                                                _Float16* __restrict__ O) {
  const int NQ = 1024, Dh = 64;
  const int bh = blockIdx.y, qblk = blockIdx.x;
  const int tid = threadIdx.x, wave = tid >> 5, lane = tid & 31;
  const int half = lane >> 4, ml = lane & 15;
  const _Float16* Qb = Q + (size_t)bh * NQ * Dh;
  const _Float16* Kb = K + (size_t)bh * NQ * Dh;
  const _Float16* Vb = V + (size_t)bh * NQ * Dh;

  __shared__ __align__(16) _Float16 Qs[128 * 72];    // stride 72 (144 B)
  __shared__ __align__(16) _Float16 Ks[32 * 72];     // [key][d]
  __shared__ __align__(16) _Float16 Vt[64 * 40];     // [d][key] (transposed)
  __shared__ __align__(16) _Float16 Ps[8][16 * 40];  // per-wave P scratch

  for (int i = tid; i < 128 * 8; i += 256) {
    const int row = i >> 3, seg = i & 7;
    async_b128(&Qs[row * 72 + seg * 8], Qb,
               (unsigned)((((qblk * 128 + row) * Dh) + seg * 8) * 2));
  }
  wait_async0();
  __syncthreads();
  const int qr = (wave * 16 + ml) * 72;
  v16h qa0 = ld_frag(&Qs[qr + half * 8], &Qs[qr + 16 + half * 8]);       // d 0..31
  v16h qa1 = ld_frag(&Qs[qr + 32 + half * 8], &Qs[qr + 48 + half * 8]);  // d 32..63
  __syncthreads();

  float m[8], l[8];
  v8f o0 = {}, o1 = {}, o2 = {}, o3 = {};
#pragma unroll
  for (int r = 0; r < 8; r++) { m[r] = -__builtin_inff(); l[r] = 0.0f; }

  for (int kb = 0; kb < NQ; kb += 32) {
    {  // stage K via async copy; V manually transposed into LDS
      const int key = tid & 31, seg = tid >> 5;
      async_b128(&Ks[key * 72 + seg * 8], Kb,
                 (unsigned)((((kb + key) * Dh) + seg * 8) * 2));
      float4 vv = *(const float4*)&Vb[(size_t)(kb + key) * Dh + seg * 8];
      const _Float16* pv = (const _Float16*)&vv;
#pragma unroll
      for (int j = 0; j < 8; j++) Vt[(seg * 8 + j) * 40 + key] = pv[j];
    }
    wait_async0();
    __syncthreads();

    // S(16x32) = Q(16x64) @ K_blkT : two accumulators, 2 chained WMMAs each
    v8f s0 = {}, s1 = {};
    {
      const int r0 = ml * 72, r1 = (16 + ml) * 72;
      v16h kb0d0 = ld_frag(&Ks[r0 + half * 8], &Ks[r0 + 16 + half * 8]);
      v16h kb0d1 = ld_frag(&Ks[r0 + 32 + half * 8], &Ks[r0 + 48 + half * 8]);
      v16h kb1d0 = ld_frag(&Ks[r1 + half * 8], &Ks[r1 + 16 + half * 8]);
      v16h kb1d1 = ld_frag(&Ks[r1 + 32 + half * 8], &Ks[r1 + 48 + half * 8]);
      s0 = wmma_f16(qa0, kb0d0, s0);
      s0 = wmma_f16(qa1, kb0d1, s0);
      s1 = wmma_f16(qa0, kb1d0, s1);
      s1 = wmma_f16(qa1, kb1d1, s1);
    }

    // online softmax; C-layout: VGPR r holds row (r + 8*half), col = ml
#pragma unroll
    for (int r = 0; r < 8; r++) {
      float a = s0[r], b2 = s1[r];
      float mx = fmaxf(a, b2);
#pragma unroll
      for (int off = 1; off < 16; off <<= 1) mx = fmaxf(mx, __shfl_xor(mx, off));
      const float mnew = fmaxf(m[r], mx);
      const float alpha = __expf(m[r] - mnew);
      const float p0 = __expf(a - mnew), p1 = __expf(b2 - mnew);
      float rs = p0 + p1;
#pragma unroll
      for (int off = 1; off < 16; off <<= 1) rs += __shfl_xor(rs, off);
      l[r] = l[r] * alpha + rs;
      m[r] = mnew;
      o0[r] *= alpha; o1[r] *= alpha; o2[r] *= alpha; o3[r] *= alpha;
      Ps[wave][(r + 8 * half) * 40 + ml] = (_Float16)p0;
      Ps[wave][(r + 8 * half) * 40 + 16 + ml] = (_Float16)p1;
    }

    // O(16x64) += P(16x32) @ V_blk(32x64); wave-private LDS round-trip for P
    v16h pa = ld_frag(&Ps[wave][ml * 40 + half * 8],
                      &Ps[wave][ml * 40 + 16 + half * 8]);
#pragma unroll
    for (int dc = 0; dc < 4; dc++) {
      const int vr = (dc * 16 + ml) * 40;
      v16h vb = ld_frag(&Vt[vr + half * 8], &Vt[vr + 16 + half * 8]);
      v8f* op = dc == 0 ? &o0 : dc == 1 ? &o1 : dc == 2 ? &o2 : &o3;
      *op = wmma_f16(pa, vb, *op);
    }
    __syncthreads();
  }

  // epilogue: O /= l, write token-major f16 (B,N,C) for the proj GEMM
  const int b = bh / 6, h = bh % 6;
#pragma unroll
  for (int r = 0; r < 8; r++) {
    const float inv = 1.0f / l[r];
    const size_t tok = (size_t)b * 1024 + qblk * 128 + wave * 16 + r + 8 * half;
    _Float16* dst = O + tok * 384 + h * 64 + ml;
    dst[0]  = (_Float16)(o0[r] * inv);
    dst[16] = (_Float16)(o1[r] * inv);
    dst[32] = (_Float16)(o2[r] * inv);
    dst[48] = (_Float16)(o3[r] * inv);
  }
}

// ----------------------------------------------- projection GEMM + bias
// Same 128x64 double-buffered async-staged WMMA structure as gemm_qkv.
__global__ __launch_bounds__(256) void gemm_proj(const _Float16* __restrict__ A,
                                                 const _Float16* __restrict__ W,
                                                 const float* __restrict__ bias,
                                                 float* __restrict__ out) {
  const int Kd = 384, Nd = 384;
  __shared__ __align__(16) _Float16 As[2][128 * 40];
  __shared__ __align__(16) _Float16 Bs[2][64 * 40];
  const int tid = threadIdx.x, wave = tid >> 5, lane = tid & 31;
  const int half = lane >> 4, ml = lane & 15;
  const int mbase = blockIdx.y * 128, nbase = blockIdx.x * 64;
  const int mg = wave >> 1, ng = wave & 1;

  auto stage = [&](int buf, int kb) {
#pragma unroll
    for (int c = 0; c < 2; c++) {
      const int idx = tid + c * 256;
      const int row = idx >> 2, seg = idx & 3;
      async_b128(&As[buf][row * 40 + seg * 8], A,
                 (unsigned)((((mbase + row) * Kd) + kb + seg * 8) * 2));
    }
    {
      const int row = tid >> 2, seg = tid & 3;
      async_b128(&Bs[buf][row * 40 + seg * 8], W,
                 (unsigned)((((nbase + row) * Kd) + kb + seg * 8) * 2));
    }
  };

  v8f acc00 = {}, acc01 = {}, acc10 = {}, acc11 = {};
  stage(0, 0);
  wait_async0();
  __syncthreads();
  int cur = 0;
  for (int kb = 0; kb < Kd; kb += 32) {
    if (kb + 32 < Kd) stage(cur ^ 1, kb + 32);
    const _Float16* A0 = &As[cur][(mg * 32 + ml) * 40];
    const _Float16* A1 = &As[cur][(mg * 32 + 16 + ml) * 40];
    const _Float16* B0 = &Bs[cur][(ng * 32 + ml) * 40];
    const _Float16* B1 = &Bs[cur][(ng * 32 + 16 + ml) * 40];
    v16h a0 = ld_frag(A0 + half * 8, A0 + 16 + half * 8);
    v16h a1 = ld_frag(A1 + half * 8, A1 + 16 + half * 8);
    v16h b0 = ld_frag(B0 + half * 8, B0 + 16 + half * 8);
    v16h b1 = ld_frag(B1 + half * 8, B1 + 16 + half * 8);
    acc00 = wmma_f16(a0, b0, acc00);
    acc01 = wmma_f16(a0, b1, acc01);
    acc10 = wmma_f16(a1, b0, acc10);
    acc11 = wmma_f16(a1, b1, acc11);
    wait_async0();
    __syncthreads();
    cur ^= 1;
  }
#pragma unroll
  for (int mi = 0; mi < 2; mi++) {
#pragma unroll
    for (int ni = 0; ni < 2; ni++) {
      v8f acc = (mi == 0) ? (ni == 0 ? acc00 : acc01)
                          : (ni == 0 ? acc10 : acc11);
      const int gcol = nbase + (ng * 2 + ni) * 16 + ml;
      const float bv = bias[gcol];
#pragma unroll
      for (int r = 0; r < 8; r++) {
        const int grow = mbase + (mg * 2 + mi) * 16 + r + 8 * half;
        out[(size_t)grow * Nd + gcol] = acc[r] + bv;
      }
    }
  }
}

// ------------------------------------------------------------- launcher
extern "C" void kernel_launch(void* const* d_in, const int* in_sizes, int n_in,
                              void* d_out, int out_size, void* d_ws,
                              size_t ws_size, hipStream_t stream) {
  (void)in_sizes; (void)n_in; (void)out_size; (void)ws_size;
  const float* x      = (const float*)d_in[0];
  const float* qkv_w  = (const float*)d_in[1];
  const float* proj_w = (const float*)d_in[2];
  const float* proj_b = (const float*)d_in[3];
  float* out = (float*)d_out;

  const int B = 8, N = 1024, C = 384, H = 6, D = 64;
  const size_t xh_n   = (size_t)B * N * C;      // 3,145,728
  const size_t wq_n   = (size_t)3 * C * C;      // 442,368
  const size_t wp_n   = (size_t)C * C;          // 147,456
  const size_t head_n = (size_t)B * H * N * D;  // 3,145,728

  char* ws = (char*)d_ws;
  size_t off = 0;
  auto take = [&](size_t elems) {
    _Float16* p = (_Float16*)(ws + off);
    off += ((elems * 2 + 255) & ~(size_t)255);
    return p;
  };
  _Float16* Xh = take(xh_n);
  _Float16* Wq = take(wq_n);
  _Float16* Wp = take(wp_n);
  _Float16* qh = take(head_n);
  _Float16* kh = take(head_n);
  _Float16* vh = take(head_n);
  _Float16* Oh = take(xh_n);

  f32_to_f16<<<(int)((xh_n + 255) / 256), 256, 0, stream>>>(x, Xh, (int)xh_n);
  f32_to_f16<<<(int)((wq_n + 255) / 256), 256, 0, stream>>>(qkv_w, Wq, (int)wq_n);
  f32_to_f16<<<(int)((wp_n + 255) / 256), 256, 0, stream>>>(proj_w, Wp, (int)wp_n);

  gemm_qkv<<<dim3(3 * C / 64, B * N / 128), 256, 0, stream>>>(Xh, Wq, qh, kh, vh);

  norm64<<<dim3(B * H * N / 8), 256, 0, stream>>>(qh, 0.125f);  // scale folded
  norm64<<<dim3(B * H * N / 8), 256, 0, stream>>>(kh, 1.0f);

  attn_fwd<<<dim3(N / 128, B * H), 256, 0, stream>>>(qh, kh, vh, Oh);

  gemm_proj<<<dim3(C / 64, B * N / 128), 256, 0, stream>>>(Oh, Wp, proj_b, out);
}
